// CausalSelfAttention_88957362634927
// MI455X (gfx1250) — matrix operand order repaired and checked by reading, not compiled, not run on hardware
//
#include <hip/hip_runtime.h>

// ---------------------------------------------------------------------------
// CDNA5 (gfx1250) causal self-attention, bf16 WMMA pipeline.
//   1) convert x, Wq, Wk, Wv fp32 -> bf16
//   2) q/k/v = x @ W^T          (v_wmma_f32_16x16x32_bf16, fp32 accum, bf16 out)
//   3) per batch: S = alpha*q@k^T (causal tile-skip, fp32 out)
//                 P = causal softmax(S) -> bf16
//                 O = P @ v     (K-loop clipped by causality, fp32 out)
// GEMM staging uses GLOBAL_LOAD_ASYNC_TO_LDS_B128 + double-buffered LDS when
// the toolchain exposes the builtins (ASYNCcnt path), else sync staging.
// ---------------------------------------------------------------------------

typedef __attribute__((ext_vector_type(16))) __bf16 v16bf;
typedef __attribute__((ext_vector_type(8)))  float  v8f;
typedef __attribute__((ext_vector_type(4)))  int    v4i;

#if defined(__AMDGCN__) && \
    __has_builtin(__builtin_amdgcn_global_load_async_to_lds_b128) && \
    __has_builtin(__builtin_amdgcn_s_wait_asynccnt)
#define USE_ASYNC_LDS 1
// Builtin prototype (from clang diagnostic): (v4i AS1*, v4i AS3*, Imm, Imm)
#define GLB128(p) ((__attribute__((address_space(1))) v4i*)(__attribute__((address_space(1))) void*)(p))
#define LDS128(p) ((__attribute__((address_space(3))) v4i*)(__attribute__((address_space(3))) void*)(p))
#else
#define USE_ASYNC_LDS 0
#endif

union FragBF {
    uint4  u[2];   // two 16B chunks loaded from LDS
    v16bf  v;      // WMMA operand view
};

// ---------------------------------------------------------------------------
// fp32 -> bf16 elementwise convert (all sizes here are multiples of 4)
// ---------------------------------------------------------------------------
__global__ __launch_bounds__(256) void f32_to_bf16_kernel(
    const float* __restrict__ in, __bf16* __restrict__ out, int n) {
    int i = (blockIdx.x * 256 + threadIdx.x) * 4;
    if (i + 3 < n) {
        float4 f = *(const float4*)(in + i);
        out[i + 0] = (__bf16)f.x;
        out[i + 1] = (__bf16)f.y;
        out[i + 2] = (__bf16)f.z;
        out[i + 3] = (__bf16)f.w;
    } else {
        for (; i < n; ++i) out[i] = (__bf16)in[i];
    }
}

// ---------------------------------------------------------------------------
// bf16 tile transpose: in [B][S][D] -> out [B][D][S]
// ---------------------------------------------------------------------------
__global__ __launch_bounds__(256) void transpose_bf16_kernel(
    const __bf16* __restrict__ in, __bf16* __restrict__ out, int S, int D) {
    __shared__ __bf16 t[32][33];
    const int d0 = blockIdx.x * 32;
    const int s0 = blockIdx.y * 32;
    const __bf16* ib = in  + (size_t)blockIdx.z * S * D;
    __bf16*       ob = out + (size_t)blockIdx.z * S * D;
    const int tid = threadIdx.x;
    const int r = tid >> 3;           // 0..31
    const int c = (tid & 7) * 4;      // 0,4,...,28
#pragma unroll
    for (int i = 0; i < 4; ++i)
        t[r][c + i] = ib[(size_t)(s0 + r) * D + d0 + c + i];
    __syncthreads();
#pragma unroll
    for (int i = 0; i < 4; ++i)
        ob[(size_t)(d0 + r) * S + s0 + c + i] = t[c + i][r];
}

// ---------------------------------------------------------------------------
// Causal row softmax: scores fp32 [S][S] (only j<=i valid) -> probs bf16,
// zeros above the diagonal. One 256-thread block per row.
// ---------------------------------------------------------------------------
__global__ __launch_bounds__(256) void softmax_rows_kernel(
    const float* __restrict__ scores, __bf16* __restrict__ probs, int S) {
    const int i   = blockIdx.x;
    const int len = i + 1;
    const float* srow = scores + (size_t)i * S;
    __bf16*      prow = probs  + (size_t)i * S;
    __shared__ float red[256];
    const int tid = threadIdx.x;

    float mx = -3.4e38f;
    for (int j = tid; j < len; j += 256) mx = fmaxf(mx, srow[j]);
    red[tid] = mx;
    __syncthreads();
    for (int s = 128; s > 0; s >>= 1) {
        if (tid < s) red[tid] = fmaxf(red[tid], red[tid + s]);
        __syncthreads();
    }
    mx = red[0];
    __syncthreads();

    float sum = 0.0f;
    for (int j = tid; j < len; j += 256) sum += expf(srow[j] - mx);
    red[tid] = sum;
    __syncthreads();
    for (int s = 128; s > 0; s >>= 1) {
        if (tid < s) red[tid] += red[tid + s];
        __syncthreads();
    }
    const float inv = 1.0f / red[0];

    for (int j = tid; j < S; j += 256) {
        float p = 0.0f;
        if (j < len) p = expf(srow[j] - mx) * inv;
        prow[j] = (__bf16)p;
    }
}

// ---------------------------------------------------------------------------
// Tiled WMMA GEMM: C[M,N] = alpha * A[M,K] @ B[N,K]^T
//   A, B: bf16 row-major.  OutT: float or __bf16.
//   CAUSAL_SKIP: skip tiles entirely above the diagonal (QK^T).
//   CAUSAL_KLIM: clip K-loop at rowTile+128 (P@V, P is lower-triangular).
// Block: 256 threads (8 waves). Tile 128x128, BK=32.
// Wave (wm=w&3, wn=w>>2) computes a 32x64 sub-tile = 2x4 WMMA 16x16 frags.
// Staging: double-buffered LDS fed by global_load_async_to_lds_b128
// (ASYNCcnt), branch-free steady state with peeled epilogue.
// ---------------------------------------------------------------------------
template <typename OutT, bool CAUSAL_SKIP, bool CAUSAL_KLIM>
__global__ __launch_bounds__(256) void gemm_bt_kernel(
    const __bf16* __restrict__ A, const __bf16* __restrict__ B,
    OutT* __restrict__ C, int M, int N, int K,
    int lda, int ldb, int ldc, float alpha) {
    const int rowTile = blockIdx.x * 128;
    const int colTile = blockIdx.y * 128;
    if (CAUSAL_SKIP && colTile >= rowTile + 128) return;  // fully masked tile
    int kEnd = K;
    if (CAUSAL_KLIM) {
        int lim = rowTile + 128;
        kEnd = (lim < K) ? lim : K;
    }

    // 128 rows x 32 bf16 (64B data) padded to 80B/row: conflict-free b128 reads.
#if USE_ASYNC_LDS
    __shared__ __align__(16) unsigned char lsA[2][128 * 80];
    __shared__ __align__(16) unsigned char lsB[2][128 * 80];
#else
    __shared__ __align__(16) unsigned char lsA[1][128 * 80];
    __shared__ __align__(16) unsigned char lsB[1][128 * 80];
#endif

    const int tid  = threadIdx.x;
    const int lane = tid & 31;
    const int w    = tid >> 5;
    const int wm   = w & 3;          // 32-row group within tile
    const int wn   = w >> 2;         // 64-col group within tile
    const int sel  = lane >> 4;      // half-wave select (ISA frag layout)
    const int l16  = lane & 15;

    // staging: each thread moves 2x16B per matrix per K-step
    const int sr = tid >> 2;         // 0..63
    const int sc = (tid & 3) * 8;    // bf16 element offset within BK row

    v8f acc[2][4];
#pragma unroll
    for (int i = 0; i < 2; ++i)
#pragma unroll
        for (int j = 0; j < 4; ++j)
            acc[i][j] = (v8f){0.f, 0.f, 0.f, 0.f, 0.f, 0.f, 0.f, 0.f};

    // Compute one BK=32 slice from LDS buffer `p`.
    auto compute_from = [&](int p) {
        // A frags (16x32 bf16, ISA 7.12.2): lane<16 -> bytes [0..15]+[32..47]
        // of its row (K0-7,K16-23); lane>=16 -> [16..31]+[48..63].
        FragBF afr[2];
#pragma unroll
        for (int fm = 0; fm < 2; ++fm) {
            const unsigned char* base = lsA[p] + (wm * 32 + fm * 16 + l16) * 80;
            afr[fm].u[0] = *(const uint4*)(base + sel * 16);
            afr[fm].u[1] = *(const uint4*)(base + 32 + sel * 16);
        }
        // B frags (32x16 bf16): lane holds column N=lane%16;
        // lanes 0-15 carry K0-7,K16-23; lanes 16-31 carry K8-15,K24-31.
        FragBF bfr[4];
#pragma unroll
        for (int fn = 0; fn < 4; ++fn) {
            const unsigned char* base = lsB[p] + (wn * 64 + fn * 16 + l16) * 80;
            bfr[fn].u[0] = *(const uint4*)(base + sel * 16);
            bfr[fn].u[1] = *(const uint4*)(base + 32 + sel * 16);
        }
#pragma unroll
        for (int fm = 0; fm < 2; ++fm)
#pragma unroll
            for (int fn = 0; fn < 4; ++fn)
                acc[fm][fn] = __builtin_amdgcn_wmma_f32_16x16x32_bf16(
                    false, afr[fm].v, false, bfr[fn].v,
                    (short)0, acc[fm][fn], false, false);
    };

#if USE_ASYNC_LDS
    // Running global pointers, advanced 32 bf16 per K-slice.
    const __bf16* ga0 = A + (size_t)(rowTile + sr)      * lda + sc;
    const __bf16* ga1 = A + (size_t)(rowTile + sr + 64) * lda + sc;
    const __bf16* gb0 = B + (size_t)(colTile + sr)      * ldb + sc;
    const __bf16* gb1 = B + (size_t)(colTile + sr + 64) * ldb + sc;

    // Issue this thread's 4 async b128 copies into LDS buffer `p` (ASYNCcnt).
    auto issue_tile = [&](int p) {
        __builtin_amdgcn_global_load_async_to_lds_b128(
            GLB128(ga0), LDS128(lsA[p] + (sr)      * 80 + sc * 2), 0, 0);
        __builtin_amdgcn_global_load_async_to_lds_b128(
            GLB128(ga1), LDS128(lsA[p] + (sr + 64) * 80 + sc * 2), 0, 0);
        __builtin_amdgcn_global_load_async_to_lds_b128(
            GLB128(gb0), LDS128(lsB[p] + (sr)      * 80 + sc * 2), 0, 0);
        __builtin_amdgcn_global_load_async_to_lds_b128(
            GLB128(gb1), LDS128(lsB[p] + (sr + 64) * 80 + sc * 2), 0, 0);
        ga0 += 32; ga1 += 32; gb0 += 32; gb1 += 32;
    };

    int p = 0;
    issue_tile(0);                              // prologue: fill buffer 0
    // Steady state (branch-free): prefetch next slice, wait for current.
    for (int kb = 32; kb < kEnd; kb += 32) {
        issue_tile(p ^ 1);
        // Async loads complete in order: <=4 outstanding => buffer p landed.
        __builtin_amdgcn_s_wait_asynccnt(4);
        __syncthreads();                        // all waves' copies visible
        compute_from(p);
        __syncthreads();                        // done reading before overwrite
        p ^= 1;
    }
    // Epilogue: last slice, nothing in flight behind it.
    __builtin_amdgcn_s_wait_asynccnt(0);
    __syncthreads();
    compute_from(p);
#else
    for (int kb = 0; kb < kEnd; kb += 32) {
        uint4 a0 = *(const uint4*)(A + (size_t)(rowTile + sr)      * lda + kb + sc);
        uint4 a1 = *(const uint4*)(A + (size_t)(rowTile + sr + 64) * lda + kb + sc);
        uint4 b0 = *(const uint4*)(B + (size_t)(colTile + sr)      * ldb + kb + sc);
        uint4 b1 = *(const uint4*)(B + (size_t)(colTile + sr + 64) * ldb + kb + sc);
        __syncthreads();
        *(uint4*)(lsA[0] + (sr)      * 80 + sc * 2) = a0;
        *(uint4*)(lsA[0] + (sr + 64) * 80 + sc * 2) = a1;
        *(uint4*)(lsB[0] + (sr)      * 80 + sc * 2) = b0;
        *(uint4*)(lsB[0] + (sr + 64) * 80 + sc * 2) = b1;
        __syncthreads();
        compute_from(0);
    }
#endif

    // C layout (ISA 7.12.2): VGPR r, lane<16 -> (M=r, N=lane);
    // lane>=16 -> (M=8+r, N=lane-16).
#pragma unroll
    for (int fm = 0; fm < 2; ++fm) {
#pragma unroll
        for (int fn = 0; fn < 4; ++fn) {
            const int col = colTile + wn * 64 + fn * 16 + l16;
#pragma unroll
            for (int r = 0; r < 8; ++r) {
                const int row = rowTile + wm * 32 + fm * 16 + sel * 8 + r;
                C[(size_t)row * ldc + col] = (OutT)(acc[fm][fn][r] * alpha);
            }
        }
    }
}

// ---------------------------------------------------------------------------
extern "C" void kernel_launch(void* const* d_in, const int* in_sizes, int n_in,
                              void* d_out, int out_size, void* d_ws, size_t ws_size,
                              hipStream_t stream) {
    (void)in_sizes; (void)n_in; (void)out_size; (void)ws_size;
    const int B = 4, S = 2048, D = 1024;
    const int M = B * S;  // 8192 token rows

    const float* x  = (const float*)d_in[0];
    const float* Wq = (const float*)d_in[1];
    const float* Wk = (const float*)d_in[2];
    const float* Wv = (const float*)d_in[3];
    float* out = (float*)d_out;

    unsigned char* ws = (unsigned char*)d_ws;
    size_t off = 0;
    auto salloc = [&](size_t bytes) {
        unsigned char* p = ws + off;
        off += (bytes + 255) & ~(size_t)255;
        return p;
    };
    __bf16* xb    = (__bf16*)salloc((size_t)M * D * 2);   // 16 MB (aliased by scores later)
    __bf16* wqb   = (__bf16*)salloc((size_t)D * D * 2);
    __bf16* wkb   = (__bf16*)salloc((size_t)D * D * 2);
    __bf16* wvb   = (__bf16*)salloc((size_t)D * D * 2);
    __bf16* qb    = (__bf16*)salloc((size_t)M * D * 2);
    __bf16* kb    = (__bf16*)salloc((size_t)M * D * 2);
    __bf16* vb    = (__bf16*)salloc((size_t)M * D * 2);
    __bf16* vT    = (__bf16*)salloc((size_t)M * D * 2);   // [B][D][S]
    __bf16* probs = (__bf16*)salloc((size_t)S * S * 2);   // per-batch, reused
    float*  scores = (float*)xb;  // 16 MB fp32, reuses xb after projections (stream-ordered)

    dim3 blk(256);

    // 1) fp32 -> bf16 conversions
    const int nx = M * D, nw = D * D;
    f32_to_bf16_kernel<<<dim3((nx / 4 + 255) / 256), blk, 0, stream>>>(x,  xb,  nx);
    f32_to_bf16_kernel<<<dim3((nw / 4 + 255) / 256), blk, 0, stream>>>(Wq, wqb, nw);
    f32_to_bf16_kernel<<<dim3((nw / 4 + 255) / 256), blk, 0, stream>>>(Wk, wkb, nw);
    f32_to_bf16_kernel<<<dim3((nw / 4 + 255) / 256), blk, 0, stream>>>(Wv, wvb, nw);

    // 2) q/k/v projections: [8192,1024] = xb @ W^T  (W stored [out,in] -> B[N,K])
    dim3 gproj(M / 128, D / 128);
    gemm_bt_kernel<__bf16, false, false><<<gproj, blk, 0, stream>>>(
        xb, wqb, qb, M, D, D, D, D, D, 1.0f);
    gemm_bt_kernel<__bf16, false, false><<<gproj, blk, 0, stream>>>(
        xb, wkb, kb, M, D, D, D, D, D, 1.0f);
    gemm_bt_kernel<__bf16, false, false><<<gproj, blk, 0, stream>>>(
        xb, wvb, vb, M, D, D, D, D, D, 1.0f);

    // 3) v -> v^T so the PV GEMM keeps contiguous-K operands
    transpose_bf16_kernel<<<dim3(D / 32, S / 32, B), blk, 0, stream>>>(vb, vT, S, D);

    const float alpha = 1.0f / sqrtf((float)D);  // softmax(scores / sqrt(d_out))
    for (int b = 0; b < B; ++b) {
        const __bf16* qbb = qb + (size_t)b * S * D;
        const __bf16* kbb = kb + (size_t)b * S * D;
        const __bf16* vtb = vT + (size_t)b * D * S;
        float*        ob  = out + (size_t)b * S * D;

        // S = alpha * q @ k^T, upper-triangle tiles skipped
        gemm_bt_kernel<float, true, false><<<dim3(S / 128, S / 128), blk, 0, stream>>>(
            qbb, kbb, scores, S, S, D, D, D, S, alpha);
        // causal softmax rows -> bf16 probabilities (zeros above diagonal)
        softmax_rows_kernel<<<dim3(S), blk, 0, stream>>>(scores, probs, S);
        // O = P @ V, K-loop clipped at rowTile+128 (P lower-triangular)
        gemm_bt_kernel<float, false, true><<<dim3(S / 128, D / 128), blk, 0, stream>>>(
            probs, vtb, ob, S, D, S, S, S, D, 1.0f);
    }
}
